// ScalarLSTM_81217831567665
// MI455X (gfx1250) — matrix operand order repaired and checked by reading, not compiled, not run on hardware
//
#include <hip/hip_runtime.h>
#include <hip/hip_bf16.h>

typedef unsigned short u16;
typedef __attribute__((ext_vector_type(8)))  float   v8f;
typedef __attribute__((ext_vector_type(4)))  float   v4f;
typedef __attribute__((ext_vector_type(16))) __bf16  v16bf;
typedef __attribute__((ext_vector_type(16))) u16     v16u;
typedef __attribute__((ext_vector_type(8)))  u16     v8u;

// ---------------------------------------------------------------------------
// helpers
// ---------------------------------------------------------------------------
__device__ __forceinline__ u16 f32_to_bf16(float f) {
  unsigned u = __float_as_uint(f);
  u += 0x7fffu + ((u >> 16) & 1u);          // round-to-nearest-even
  return (u16)(u >> 16);
}

__device__ __forceinline__ v16bf make_frag(const u16* p0, const u16* p1) {
  v8u lo = *(const v8u*)p0;
  v8u hi = *(const v8u*)p1;
  v16u t;
#pragma unroll
  for (int i = 0; i < 8; i++) { t[i] = lo[i]; t[i + 8] = hi[i]; }
  union { v16u u; v16bf b; } cv;
  cv.u = t;
  return cv.b;
}

// ---------------------------------------------------------------------------
// 1) precision convert / weight repack
// ---------------------------------------------------------------------------
__global__ void convert_kernel(const float* __restrict__ x,
                               const float* __restrict__ Wq, const float* __restrict__ Wk,
                               const float* __restrict__ Wv, const float* __restrict__ Wf,
                               const float* __restrict__ Wi, const float* __restrict__ Wo,
                               const float* __restrict__ Wout,
                               const float* __restrict__ bq, const float* __restrict__ bk,
                               const float* __restrict__ bv, const float* __restrict__ bf_,
                               const float* __restrict__ bi, const float* __restrict__ bo,
                               u16* __restrict__ xbf, u16* __restrict__ wcat,
                               u16* __restrict__ woutbf, float* __restrict__ biascat) {
  const int stride = gridDim.x * blockDim.x;
  const int gid = blockIdx.x * blockDim.x + threadIdx.x;
  const float* Ws[6] = {Wq, Wk, Wv, Wf, Wi, Wo};
  const float* bs[6] = {bq, bk, bv, bf_, bi, bo};

  for (int i = gid; i < 4096 * 256; i += stride) xbf[i] = f32_to_bf16(x[i]);
  for (int i = gid; i < 1536 * 256; i += stride) {
    int row = i >> 8, k = i & 255;            // row in 0..1535
    int g = row >> 8, r = row & 255;
    wcat[i] = f32_to_bf16(Ws[g][r * 512 + k]); // x-part = cols 0..255
  }
  for (int i = gid; i < 256 * 256; i += stride) woutbf[i] = f32_to_bf16(Wout[i]);
  for (int i = gid; i < 1536; i += stride) biascat[i] = bs[i >> 8][i & 255];
}

// ---------------------------------------------------------------------------
// 2/4) WMMA GEMM: Out[M,N] = A[M,256]bf16 * Bw[N,256]bf16^T + bias
//     block = 256 threads (8 waves), WG tile 64x32, wave tile 16x16, K = 8x32
// ---------------------------------------------------------------------------
__global__ __launch_bounds__(256) void wmma_gemm_kernel(const u16* __restrict__ A,
                                                        const u16* __restrict__ Bw,
                                                        const float* __restrict__ bias,
                                                        float* __restrict__ Out,
                                                        int Nstride) {
  const int tid = threadIdx.x;
  const int wave = tid >> 5;
  const int lane = tid & 31;
  const int half = lane >> 4;
  const int lr = lane & 15;
  const int m_base = blockIdx.x * 64 + (wave & 3) * 16;
  const int n_base = blockIdx.y * 32 + (wave >> 2) * 16;
  const int nn = n_base + lr;

  float bv = bias[nn];
  v8f acc;
#pragma unroll
  for (int i = 0; i < 8; i++) acc[i] = bv;

  const u16* arow = A + (long)(m_base + lr) * 256;  // both lane-halves share rows
  const u16* brow = Bw + (long)nn * 256;            // lane -> output column nn

#pragma unroll
  for (int kc = 0; kc < 8; kc++) {
    const int k0 = kc * 32;
    v16bf a = make_frag(arow + k0 + half * 8, arow + k0 + 16 + half * 8);
    v16bf b = make_frag(brow + k0 + half * 8, brow + k0 + 16 + half * 8);
    acc = __builtin_amdgcn_wmma_f32_16x16x32_bf16(false, a, false, b,
                                                  (short)0, acc, false, false);
  }
#pragma unroll
  for (int v = 0; v < 8; v++) {
    int m = m_base + v + half * 8;
    Out[(long)m * Nstride + nn] = acc[v];
  }
}

// ---------------------------------------------------------------------------
// 3) recurrence: one workgroup per batch item, C (256KB) resident in LDS.
//    3 barriers per step; den/num reductions use wave32 shuffles.
// ---------------------------------------------------------------------------
__global__ __launch_bounds__(512) void recurrent_kernel(
    const float* __restrict__ G,   // [B,T,1536] precomputed x-projections + bias
    const float* __restrict__ Wq, const float* __restrict__ Wk,
    const float* __restrict__ Wv, const float* __restrict__ Wf,
    const float* __restrict__ Wi, const float* __restrict__ Wo,
    u16* __restrict__ Hseq,        // [B,T,256] bf16 for the post-GEMM
    float* __restrict__ out_base)  // d_out (final states region)
{
  __shared__ float C_s[256 * 256];      // 256 KB state (CDNA5: 320 KB LDS/WGP)
  __shared__ float h_s[256], n_s[256], m_s[256];
  __shared__ float g_s[1536];           // q|k|v|f|i|o at 0,256,512,768,1024,1280
  __shared__ float ft_s[256], it_s[256];
  __shared__ float denp_s[8];           // per-wave den partials

  const int b = blockIdx.x;
  const int tid = threadIdx.x;

  for (int i = tid; i < 65536; i += 512) C_s[i] = 0.f;
  if (tid < 256) { h_s[tid] = 0.f; n_s[tid] = 0.f; m_s[tid] = 0.f; }
  __syncthreads();

  const float* Ws[6] = {Wq, Wk, Wv, Wf, Wi, Wo};
  const int o0 = tid * 3;               // 512 threads x 3 outputs = 1536 gates
  const float* w0 = Ws[(o0    ) >> 8] + ((o0    ) & 255) * 512 + 256; // h-part cols
  const float* w1 = Ws[(o0 + 1) >> 8] + ((o0 + 1) & 255) * 512 + 256;
  const float* w2 = Ws[(o0 + 2) >> 8] + ((o0 + 2) & 255) * 512 + 256;
  const float inv_sqrt_h = 0.0625f;     // 1/sqrt(256)

  for (int t = 0; t < 512; t++) {
    // ---- gate matvec: g = G[b,t] + W_h @ h (float4 LDS + global b128) ----
    const float* Gt = G + ((long)(b * 512 + t)) * 1536;
    float a0 = Gt[o0], a1 = Gt[o0 + 1], a2 = Gt[o0 + 2];
#pragma unroll 2
    for (int k = 0; k < 256; k += 4) {
      v4f hv = *(const v4f*)&h_s[k];
      v4f x0 = *(const v4f*)&w0[k];
      v4f x1 = *(const v4f*)&w1[k];
      v4f x2 = *(const v4f*)&w2[k];
#pragma unroll
      for (int j = 0; j < 4; j++) {
        a0 = __fmaf_rn(x0[j], hv[j], a0);
        a1 = __fmaf_rn(x1[j], hv[j], a1);
        a2 = __fmaf_rn(x2[j], hv[j], a2);
      }
    }
    g_s[o0] = a0; g_s[o0 + 1] = a1; g_s[o0 + 2] = a2;
    __syncthreads();                                   // B1: g ready

    // ---- stabilized gates, n update, wave-level den partials -------------
    if (tid < 256) {
      float fl = g_s[768 + tid];
      float il = g_s[1024 + tid];
      float mprev = m_s[tid];
      float mn = fmaxf(fl + mprev, il);
      float it = __expf(il - mn);
      float ft = __expf(fl + mprev - mn);
      m_s[tid] = mn; ft_s[tid] = ft; it_s[tid] = it;
      float kk = g_s[256 + tid] * inv_sqrt_h;
      g_s[256 + tid] = kk;
      float og = g_s[1280 + tid];
      g_s[1280 + tid] = 1.f / (1.f + __expf(-og));     // sigmoid(o)
      float nnew = __fmaf_rn(ft, n_s[tid], it * kk);
      n_s[tid] = nnew;
      float part = nnew * g_s[tid];                    // n_new * q
#pragma unroll
      for (int off = 16; off > 0; off >>= 1)           // wave32 reduction
        part += __shfl_xor(part, off, 32);
      if ((tid & 31) == 0) denp_s[tid >> 5] = part;
    }
    __syncthreads();                                   // B2: gates + denp ready

    // ---- C update fused with num = C_new @ q, h via lane-pair shuffle ----
    {
      const int row = tid >> 1;
      const int ch = tid & 1;
      const float ft = ft_s[row], it = it_s[row], vv = g_s[512 + row];
      const float ivv = it * vv;
      float acc = 0.f;
      float* Crow = C_s + row * 256 + ch * 128;
      const float* kks = g_s + 256 + ch * 128;
      const float* qs = g_s + ch * 128;
#pragma unroll 4
      for (int c = 0; c < 128; c++) {
        float Cv = __fmaf_rn(ft, Crow[c], ivv * kks[c]);
        Crow[c] = Cv;
        acc = __fmaf_rn(Cv, qs[c], acc);
      }
      float num = acc + __shfl_xor(acc, 1, 32);        // combine the two half-rows
      if (ch == 0) {
        float den = fabsf(denp_s[0] + denp_s[1] + denp_s[2] + denp_s[3] +
                          denp_s[4] + denp_s[5] + denp_s[6] + denp_s[7]) + 1e-8f;
        float hnew = g_s[1280 + row] * (num / den);
        h_s[row] = hnew;
        Hseq[((long)(b * 512 + t)) * 256 + row] = f32_to_bf16(hnew);
      }
    }
    __syncthreads();                                   // B3: h ready for next step
  }

  // ---- final state tuple (h, C, n, m) into d_out -------------------------
  const long OFF_H = 1048576, OFF_C = 1050624, OFF_N = 1574912, OFF_M = 1576960;
  for (int i = tid; i < 65536; i += 512)
    out_base[OFF_C + (long)b * 65536 + i] = C_s[i];
  if (tid < 256) {
    out_base[OFF_H + b * 256 + tid] = h_s[tid];
    out_base[OFF_N + b * 256 + tid] = n_s[tid];
    out_base[OFF_M + b * 256 + tid] = m_s[tid];
  }
}

// ---------------------------------------------------------------------------
// launch
// ---------------------------------------------------------------------------
extern "C" void kernel_launch(void* const* d_in, const int* in_sizes, int n_in,
                              void* d_out, int out_size, void* d_ws, size_t ws_size,
                              hipStream_t stream) {
  const float* x    = (const float*)d_in[0];
  const float* Wq   = (const float*)d_in[1];  const float* bq   = (const float*)d_in[2];
  const float* Wk   = (const float*)d_in[3];  const float* bk   = (const float*)d_in[4];
  const float* Wv   = (const float*)d_in[5];  const float* bv   = (const float*)d_in[6];
  const float* Wf   = (const float*)d_in[7];  const float* bf_  = (const float*)d_in[8];
  const float* Wi   = (const float*)d_in[9];  const float* bi   = (const float*)d_in[10];
  const float* Wo   = (const float*)d_in[11]; const float* bo   = (const float*)d_in[12];
  const float* Wout = (const float*)d_in[13]; const float* bout = (const float*)d_in[14];

  char* ws = (char*)d_ws;
  u16*   xbf     = (u16*)(ws + 0);          //  2 MB : x bf16 [4096,256]
  u16*   wcat    = (u16*)(ws + 2097152);    // 768 KB: gate x-weights bf16 [1536,256]
  u16*   woutbf  = (u16*)(ws + 2883584);    // 128 KB: Wout bf16 [256,256]
  float* biascat = (float*)(ws + 3014656);  //   6 KB: gate biases [1536]
  float* G       = (float*)(ws + 3020800);  //  24 MB: [B,T,1536] f32
  u16*   hseq    = (u16*)(ws + 28186624);   //   2 MB: h sequence bf16 [4096,256]
  float* out = (float*)d_out;

  convert_kernel<<<1024, 256, 0, stream>>>(x, Wq, Wk, Wv, Wf, Wi, Wo, Wout,
                                           bq, bk, bv, bf_, bi, bo,
                                           xbf, wcat, woutbf, biascat);

  // G = x @ Wcat^T + bias  (M=4096, N=1536, K=256)
  wmma_gemm_kernel<<<dim3(64, 48), 256, 0, stream>>>(xbf, wcat, biascat, G, 1536);

  // sequential mLSTM recurrence, one WGP per batch item, C in LDS
  recurrent_kernel<<<8, 512, 0, stream>>>(G, Wq, Wk, Wv, Wf, Wi, Wo, hseq, out);

  // outputs = Hseq @ Wout^T + bout  (M=4096, N=256, K=256)
  wmma_gemm_kernel<<<dim3(64, 16), 256, 0, stream>>>(hseq, woutbf, bout, out, 256);
}
